// SlidingWindowAttention_75917841924763
// MI455X (gfx1250) — matrix operand order repaired and checked by reading, not compiled, not run on hardware
//
#include <hip/hip_runtime.h>
#include <cstdint>
#include <cstddef>

// ---------------------------------------------------------------------------
// Sliding-window attention (B=2, T=2048, C=1024, H=16, D=64, W=256) for
// gfx1250: all matmuls on V_WMMA_F32_16X16X32_F16 (f32 accum), f16 data,
// double-buffered LDS staging via GLOBAL_LOAD_ASYNC_TO_LDS when available.
// ---------------------------------------------------------------------------

typedef _Float16 h4   __attribute__((ext_vector_type(4)));
typedef _Float16 h8   __attribute__((ext_vector_type(8)));
typedef _Float16 v16h __attribute__((ext_vector_type(16)));
typedef float    v8f  __attribute__((ext_vector_type(8)));
typedef int      v4i  __attribute__((ext_vector_type(4)));

union FragU { v16h v; h8 h[2]; };

#if defined(__AMDGCN__) && __has_builtin(__builtin_amdgcn_global_load_async_to_lds_b128) && \
    __has_builtin(__builtin_amdgcn_s_wait_asynccnt)
#define USE_ASYNC_LDS 1
#endif

// Copy 8 f16 (16B) global -> LDS. Async (ASYNCcnt-tracked) when available.
__device__ __forceinline__ void cp16_g2l(const _Float16* g, _Float16* l) {
#ifdef USE_ASYNC_LDS
  __builtin_amdgcn_global_load_async_to_lds_b128(
      (__attribute__((address_space(1))) v4i*)g,
      (__attribute__((address_space(3))) v4i*)l, 0, 0);
#else
  *(h8*)l = *(const h8*)g;
#endif
}

#ifdef USE_ASYNC_LDS
#define ASYNC_WAIT(n) __builtin_amdgcn_s_wait_asynccnt(n)
#else
#define ASYNC_WAIT(n)
#endif

// A-fragment (16x32, 16-bit): lane = m (lane&15); lanes 0-15 hold K
// {0..7,16..23}, lanes 16-31 hold K {8..15,24..31}: two 16B LDS reads.
__device__ __forceinline__ v16h load_a_frag(const _Float16* rowPtr, int kBase, int hi) {
  FragU f;
  f.h[0] = *(const h8*)(rowPtr + kBase + hi * 8);
  f.h[1] = *(const h8*)(rowPtr + kBase + 16 + hi * 8);
  return f.v;
}

// B-fragment (32x16, 16-bit): lane = n (lane&15); elements = K in the same
// order as A ({0..7,16..23} / {8..15,24..31}), strided column gather.
__device__ __forceinline__ v16h load_b_frag(const _Float16* p, int st, int hi) {
  v16h v;
  for (int s = 0; s < 16; ++s) v[s] = p[(s < 8 ? 8 * hi + s : 16 + 8 * hi + s - 8) * st];
  return v;
}

__device__ __forceinline__ v8f wmma_f16(v16h a, v16h b, v8f c) {
  return __builtin_amdgcn_wmma_f32_16x16x32_f16(false, a, false, b, (short)0, c,
                                                false, false);
}

// ---------------------------------------------------------------------------
// f32 -> f16 elementwise convert (n4 = n/4 vector elements).
// ---------------------------------------------------------------------------
__global__ __launch_bounds__(256) void cvt_f16_kernel(const float* __restrict__ src,
                                                      _Float16* __restrict__ dst,
                                                      int n4) {
  int i = blockIdx.x * 256 + threadIdx.x;
  if (i < n4) {
    float4 v = ((const float4*)src)[i];
    h4 o = {(_Float16)v.x, (_Float16)v.y, (_Float16)v.z, (_Float16)v.w};
    ((h4*)dst)[i] = o;
  }
}

// ---------------------------------------------------------------------------
// W[N][K] (f32) -> Wt[K][N] (f16) tiled transpose.
// ---------------------------------------------------------------------------
__global__ __launch_bounds__(256) void transpose_cvt_kernel(const float* __restrict__ W,
                                                            _Float16* __restrict__ Wt,
                                                            int N, int K) {
  __shared__ _Float16 tile[32][33];
  const int bk = blockIdx.x * 32;  // k tile
  const int bn = blockIdx.y * 32;  // n tile
  const int tx = threadIdx.x & 31, ty = threadIdx.x >> 5;
  for (int r = ty; r < 32; r += 8)
    tile[r][tx] = (_Float16)W[(size_t)(bn + r) * K + bk + tx];
  __syncthreads();
  for (int r = ty; r < 32; r += 8)
    Wt[(size_t)(bk + r) * N + bn + tx] = tile[tx][r];
}

// ---------------------------------------------------------------------------
// Y[M,N] = X[M,K](f16) * Wt[K,N](f16) + bias[N]
// 256 threads (8 waves); tile 128(M) x 128(N); K step 32; double-buffered LDS.
// Wave (wm,wn) owns 32(m) x 64(n) = 2x4 WMMA tiles.
// ---------------------------------------------------------------------------
template <typename TOut>
__global__ __launch_bounds__(256) void gemm_wmma(const _Float16* __restrict__ X,
                                                 const _Float16* __restrict__ Wt,
                                                 const float* __restrict__ bias,
                                                 TOut* __restrict__ Y,
                                                 int M, int N, int K) {
  constexpr int BM = 128, BN = 128, BK = 32;
  constexpr int AST = 40;   // A row stride (elems): 80B, 16B-aligned, bank-spread
  constexpr int BST = 136;  // B row stride (elems): 272B, 16B-aligned, bank-spread
  __shared__ _Float16 As[2][BM][AST];  // [buf][m][k]
  __shared__ _Float16 Bs[2][BK][BST];  // [buf][k][n]

  const int t = threadIdx.x;
  const int lane = t & 31, wave = t >> 5;
  const int wm = wave & 3, wn = wave >> 2;
  const int hi = lane >> 4, ln = lane & 15;
  const int m0 = blockIdx.y * BM, n0 = blockIdx.x * BN;

  v8f acc[2][4];
  for (int j = 0; j < 4; ++j) {
    float bv = bias[n0 + wn * 64 + j * 16 + ln];
    for (int i = 0; i < 2; ++i)
      for (int r = 0; r < 8; ++r) acc[i][j][r] = bv;
  }

  // Stage one (A,B) tile pair: 4 B128 copies per thread.
  auto stage = [&](int buf, int kk) {
    for (int i = 0; i < 2; ++i) {  // A: 128 rows x 32 halves = 512 B128
      int idx = t + 256 * i;
      int r = idx >> 2, sg = idx & 3;
      cp16_g2l(&X[(size_t)(m0 + r) * K + kk + sg * 8], &As[buf][r][sg * 8]);
    }
    for (int i = 0; i < 2; ++i) {  // B: 32 rows x 128 halves = 512 B128
      int idx = t + 256 * i;
      int kr = idx >> 4, sg = idx & 15;
      cp16_g2l(&Wt[(size_t)(kk + kr) * N + n0 + sg * 8], &Bs[buf][kr][sg * 8]);
    }
  };

  stage(0, 0);
  int buf = 0;
  for (int kk = 0; kk < K; kk += BK) {
    __syncthreads();  // all waves done computing from buf^1 before overwrite
    if (kk + BK < K) {
      stage(buf ^ 1, kk + BK);
      ASYNC_WAIT(4);  // >=4 newest outstanding are next-stage; current done
    } else {
      ASYNC_WAIT(0);
    }
    __syncthreads();  // current buffer visible from all waves

    v16h a0 = load_a_frag(&As[buf][wm * 32 + ln][0], 0, hi);
    v16h a1 = load_a_frag(&As[buf][wm * 32 + 16 + ln][0], 0, hi);
    v16h b[4];
    for (int j = 0; j < 4; ++j) b[j] = load_b_frag(&Bs[buf][0][wn * 64 + j * 16 + ln], BST, hi);
    for (int j = 0; j < 4; ++j) {
      acc[0][j] = wmma_f16(a0, b[j], acc[0][j]);
      acc[1][j] = wmma_f16(a1, b[j], acc[1][j]);
    }
    buf ^= 1;
  }

  // C layout: lane -> n = ln; VGPR r -> m = r + hi*8.
  for (int i = 0; i < 2; ++i)
    for (int j = 0; j < 4; ++j)
      for (int r = 0; r < 8; ++r) {
        int row = m0 + wm * 32 + i * 16 + hi * 8 + r;
        int col = n0 + wn * 64 + j * 16 + ln;
        Y[(size_t)row * N + col] = (TOut)acc[i][j][r];
      }
}

// ---------------------------------------------------------------------------
// Sliding-window flash attention. One workgroup per (64-query block, head, b).
// 128 threads = 4 waves; wave w owns query rows [w*16, w*16+16).
// Key chunks of 64; at most 5 chunks (window 256 + causal diagonal).
// ---------------------------------------------------------------------------
__global__ __launch_bounds__(128) void attn_kernel(const _Float16* __restrict__ Qh,
                                                   const _Float16* __restrict__ Kh,
                                                   const _Float16* __restrict__ Vh,
                                                   _Float16* __restrict__ AO) {
  constexpr int T = 2048, C = 1024, DH = 64, WND = 256;
  constexpr int ST = 72;
  __shared__ _Float16 Qs[64][ST];   // [q][d]
  __shared__ _Float16 KT[64][ST];   // [d][key]
  __shared__ _Float16 Vs[64][ST];   // [key][d]
  __shared__ _Float16 Ps[64][ST];   // [q][key] probs f16
  __shared__ float    Ss[64][66];   // [q][key] scores f32
  __shared__ float    mI[64], lI[64], scI[64];

  const int t = threadIdx.x;
  const int lane = t & 31, wave = t >> 5;
  const int hi = lane >> 4, ln = lane & 15;
  const int q0 = blockIdx.x * 64;
  const int h = blockIdx.y, b = blockIdx.z;
  const size_t base = ((size_t)b * T) * C + (size_t)h * DH;

  for (int i = 0; i < 8; ++i) {  // Q tile 64x64, B128 loads
    int idx = t + 128 * i;
    int r = idx >> 3, sg = idx & 7;
    *(h8*)&Qs[r][sg * 8] = *(const h8*)&Qh[base + (size_t)(q0 + r) * C + sg * 8];
  }
  if (t < 64) { mI[t] = -__builtin_inff(); lI[t] = 0.0f; }

  v8f o[4];
  for (int nt = 0; nt < 4; ++nt)
    for (int r = 0; r < 8; ++r) o[nt][r] = 0.0f;

  const int c0lo = (q0 >= WND) ? (q0 - WND) : 0;
  for (int c0 = c0lo; c0 <= q0; c0 += 64) {
    __syncthreads();
    // Stage K (transposed) and V chunks (64 keys x 64 dims).
    for (int i = 0; i < 32; ++i) {
      int idx = t + 128 * i;
      int r = idx >> 6, c = idx & 63;
      KT[c][r] = Kh[base + (size_t)(c0 + r) * C + c];
      Vs[r][c] = Vh[base + (size_t)(c0 + r) * C + c];
    }
    // Prefetch next chunk into cache (global_prefetch_b8).
    if (c0 + 64 <= q0) {
      __builtin_prefetch(&Kh[base + (size_t)(c0 + 64 + (t & 63)) * C], 0, 0);
      __builtin_prefetch(&Vh[base + (size_t)(c0 + 64 + (t & 63)) * C], 0, 0);
    }
    __syncthreads();

    // S = Q K^T / sqrt(64); wave's 16 rows x 64 keys -> LDS (f32).
    const _Float16* qrow = &Qs[wave * 16 + ln][0];
    for (int nt = 0; nt < 4; ++nt) {
      v8f s;
      for (int r = 0; r < 8; ++r) s[r] = 0.0f;
      s = wmma_f16(load_a_frag(qrow, 0, hi), load_b_frag(&KT[0][nt * 16 + ln], ST, hi), s);
      s = wmma_f16(load_a_frag(qrow, 32, hi), load_b_frag(&KT[32][nt * 16 + ln], ST, hi), s);
      for (int r = 0; r < 8; ++r)
        Ss[wave * 16 + hi * 8 + r][nt * 16 + ln] = s[r] * 0.125f;
    }
    __syncthreads();

    // Online softmax: thread t (<64) owns query row q0+t.
    if (t < 64) {
      const int q = q0 + t;
      const float mo = mI[t];
      float mx = mo;
      for (int j = 0; j < 64; ++j) {
        int key = c0 + j;
        if (key <= q && (q - key) < WND) mx = fmaxf(mx, Ss[t][j]);
      }
      float scale, sum = 0.0f;
      if (mx == -__builtin_inff()) {  // fully masked chunk for this row
        scale = 1.0f;
        for (int j = 0; j < 64; ++j) Ps[t][j] = (_Float16)0.0f;
      } else {
        scale = __expf(mo - mx);
        for (int j = 0; j < 64; ++j) {
          int key = c0 + j;
          float p = 0.0f;
          if (key <= q && (q - key) < WND) p = __expf(Ss[t][j] - mx);
          Ps[t][j] = (_Float16)p;
          sum += p;
        }
      }
      lI[t] = lI[t] * scale + sum;
      mI[t] = mx;
      scI[t] = scale;
    }
    __syncthreads();

    // Rescale O and accumulate O += P * V.
    float sc[8];
    for (int r = 0; r < 8; ++r) sc[r] = scI[wave * 16 + hi * 8 + r];
    const _Float16* prow = &Ps[wave * 16 + ln][0];
    for (int nt = 0; nt < 4; ++nt) {
      for (int r = 0; r < 8; ++r) o[nt][r] *= sc[r];
      o[nt] = wmma_f16(load_a_frag(prow, 0, hi), load_b_frag(&Vs[0][nt * 16 + ln], ST, hi), o[nt]);
      o[nt] = wmma_f16(load_a_frag(prow, 32, hi), load_b_frag(&Vs[32][nt * 16 + ln], ST, hi), o[nt]);
    }
  }

  float linv[8];
  for (int r = 0; r < 8; ++r) linv[r] = 1.0f / lI[wave * 16 + hi * 8 + r];
  for (int nt = 0; nt < 4; ++nt)
    for (int r = 0; r < 8; ++r) {
      int row = q0 + wave * 16 + hi * 8 + r;
      AO[base + (size_t)row * C + nt * 16 + ln] = (_Float16)(o[nt][r] * linv[r]);
    }
}

// ---------------------------------------------------------------------------
// Host launcher. Inputs: x, Wq, bq, Wk, bk, Wv, bv, Wo, bo (f32).
// Workspace (f16): Qh,Kh,Vh,AO (4M each), xh (4M), Wqt,Wkt,Wvt,Wot (1M each)
// = 24M f16 = 48 MiB.
// ---------------------------------------------------------------------------
extern "C" void kernel_launch(void* const* d_in, const int* in_sizes, int n_in,
                              void* d_out, int out_size, void* d_ws, size_t ws_size,
                              hipStream_t stream) {
  constexpr int Bv = 2, Tv = 2048, Cv = 1024;
  constexpr int Mv = Bv * Tv;  // 4096
  constexpr size_t XE = (size_t)Mv * Cv;  // 4M elements
  constexpr size_t WE = (size_t)Cv * Cv;  // 1M elements

  const float* x  = (const float*)d_in[0];
  const float* Wq = (const float*)d_in[1];
  const float* bq = (const float*)d_in[2];
  const float* Wk = (const float*)d_in[3];
  const float* bk = (const float*)d_in[4];
  const float* Wv = (const float*)d_in[5];
  const float* bv = (const float*)d_in[6];
  const float* Wo = (const float*)d_in[7];
  const float* bo = (const float*)d_in[8];

  _Float16* Qh  = (_Float16*)d_ws;
  _Float16* Kh  = Qh + XE;
  _Float16* Vh  = Kh + XE;
  _Float16* AO  = Vh + XE;
  _Float16* xh  = AO + XE;
  _Float16* Wqt = xh + XE;
  _Float16* Wkt = Wqt + WE;
  _Float16* Wvt = Wkt + WE;
  _Float16* Wot = Wvt + WE;

  // Precision pre-pass: x -> f16, W* -> f16 transposed [K][N].
  cvt_f16_kernel<<<dim3((XE / 4 + 255) / 256), 256, 0, stream>>>(x, xh, (int)(XE / 4));
  dim3 gT(Cv / 32, Cv / 32);
  transpose_cvt_kernel<<<gT, 256, 0, stream>>>(Wq, Wqt, Cv, Cv);
  transpose_cvt_kernel<<<gT, 256, 0, stream>>>(Wk, Wkt, Cv, Cv);
  transpose_cvt_kernel<<<gT, 256, 0, stream>>>(Wv, Wvt, Cv, Cv);
  transpose_cvt_kernel<<<gT, 256, 0, stream>>>(Wo, Wot, Cv, Cv);

  dim3 gGemm(Cv / 128, Mv / 128);  // (8, 32)
  gemm_wmma<_Float16><<<gGemm, 256, 0, stream>>>(xh, Wqt, bq, Qh, Mv, Cv, Cv);
  gemm_wmma<_Float16><<<gGemm, 256, 0, stream>>>(xh, Wkt, bk, Kh, Mv, Cv, Cv);
  gemm_wmma<_Float16><<<gGemm, 256, 0, stream>>>(xh, Wvt, bv, Vh, Mv, Cv, Cv);

  attn_kernel<<<dim3(Tv / 64, 16, Bv), 128, 0, stream>>>(Qh, Kh, Vh, AO);

  gemm_wmma<float><<<gGemm, 256, 0, stream>>>(AO, Wot, bo, (float*)d_out, Mv, Cv, Cv);
}
